// NeuronCircuit_25864293056820
// MI455X (gfx1250) — matrix operand order repaired and checked by reading, not compiled, not run on hardware
//
#include <hip/hip_runtime.h>
#include <hip/hip_bf16.h>

// ---------------------------------------------------------------------------
// NeuronCircuit on MI455X (gfx1250, wave32, WMMA bf16 16x16x32 f32-acc)
//
//   h     = [Xscaled(cw)] @ CN_flat        (K = N*D = 16384)
//   Q/K/V = [Hscaled(ew)] @ P_flat         (K = N*R = 2048) -> [B][H][S][DH] bf16
//   causal flash attention, TDM-staged K/V tiles, 32-key blocks
//   Y     = AO @ W_O^T                     (K = D = 1024)   -> fp32 d_out
//
// GEMM waves compute 32x64 tiles (8 accumulators). K-loop ping-pongs two
// fragment sets (no rotation copies, no spills; launch_bounds gives the
// allocator a full VGPR budget). Attention stages K/V tiles with
// tensor_load_to_lds, double buffered on TENSORcnt. Workspace ~48 MB.
// ---------------------------------------------------------------------------

#define DEV __device__ __forceinline__

static constexpr int Bn  = 2;
static constexpr int Sn  = 2048;
static constexpr int Dn  = 1024;
static constexpr int Nn  = 16;
static constexpr int Rn  = 128;
static constexpr int Hn  = 16;
static constexpr int DHn = 64;

typedef __bf16          v16bf __attribute__((ext_vector_type(16)));
typedef float           v8f   __attribute__((ext_vector_type(8)));
typedef unsigned short  v16us __attribute__((ext_vector_type(16)));
typedef unsigned int    u32x4 __attribute__((ext_vector_type(4)));
typedef int             i32x8 __attribute__((ext_vector_type(8)));
typedef int             i32x4 __attribute__((ext_vector_type(4)));

#if defined(__has_builtin)
#  if __has_builtin(__builtin_amdgcn_tensor_load_to_lds) && \
      __has_builtin(__builtin_amdgcn_s_wait_tensorcnt)
#    define HAVE_TDM 1
#  endif
#endif
#ifndef HAVE_TDM
#  define HAVE_TDM 0
#endif

union BFrag { v16us u; v16bf b; };

struct Frags { BFrag A0, A1, B[4]; };     // one k-step worth of operands

DEV unsigned short bfb(float f) {          // fp32 -> bf16 bits
    union { __bf16 h; unsigned short u; } c;
    c.h = (__bf16)f;
    return c.u;
}

DEV v8f wmma_bf16(const BFrag& a, const BFrag& b, v8f c) {
    return __builtin_amdgcn_wmma_f32_16x16x32_bf16(false, a.b, false, b.b,
                                                   (short)0, c, false, false);
}

DEV void wmma8(const Frags& F, v8f acc[2][4]) {
#pragma unroll
    for (int t = 0; t < 4; ++t) {
        acc[0][t] = wmma_bf16(F.A0, F.B[t], acc[0][t]);
        acc[1][t] = wmma_bf16(F.A1, F.B[t], acc[1][t]);
    }
}

// A fragment (16x32): row = lane%16; K = half*16+0..15 (contiguous)
DEV void load_a_f32(BFrag& A, const float* rowp, int kb, int half, float scale) {
    const float* p0 = rowp + kb + half * 16;
    const float* p1 = p0 + 8;
#pragma unroll
    for (int e = 0; e < 8; ++e) {
        A.b[e]     = (__bf16)(p0[e] * scale);
        A.b[e + 8] = (__bf16)(p1[e] * scale);
    }
}

DEV void load_a_bf(BFrag& A, const unsigned short* rowp, int kb, int half) {
    const unsigned short* p0 = rowp + kb + half * 16;
    const unsigned short* p1 = p0 + 8;
#pragma unroll
    for (int e = 0; e < 8; ++e) { A.u[e] = p0[e]; A.u[e + 8] = p1[e]; }
}

// B fragment: col = lane%16, K = half*16+0..15, from B^T-contiguous memory (32B aligned)
DEV void load_b_contig(BFrag& B, const unsigned short* p) {
    B.u = *(const v16us*)p;
}

// ---------------------------------------------------------------------------
// TDM: 2D bf16 tile load (tile_dim0=64 contiguous elems, tile_dim1=32 rows,
// row stride 64 elems) global -> LDS.  D# per CDNA5 ISA.
// 6-arg builtin on this toolchain: (u32x4, i32x8, i32x4, i32x4, i32x8, cpol)
// ---------------------------------------------------------------------------
DEV void tdm_load_kv(const unsigned short* gsrc, unsigned short* ldst) {
#if HAVE_TDM
    unsigned long long ga = (unsigned long long)(size_t)gsrc;
    unsigned lds_off = (unsigned)(size_t)ldst;           // low 32 bits = LDS byte addr
    u32x4 g0 = { 1u,                                     // count=1, user descriptor
                 lds_off,
                 (unsigned)ga,
                 (unsigned)((ga >> 32) & 0x1FFFFFFu) | 0x80000000u }; // type=2
    i32x8 g1 = { (int)(1u << 16),          // data_size = 2 bytes
                 (int)(64u << 16),         // tensor_dim0 = 64  (bits 79:48)
                 (int)(2048u << 16),       // tensor_dim1 = 2048 (bits 111:80)
                 (int)(64u << 16),         // tile_dim0 = 64    (bits 127:112)
                 32,                       // tile_dim1 = 32, tile_dim2 = 0
                 64,                       // tensor_dim0_stride = 64 (bits 207:160)
                 0, 0 };                   // tensor_dim1_stride = 0
    i32x4 z4 = { 0, 0, 0, 0 };
    i32x8 z8 = { 0, 0, 0, 0, 0, 0, 0, 0 };
    __builtin_amdgcn_tensor_load_to_lds(g0, g1, z4, z4, z8, 0);
#else
    const int lane = threadIdx.x & 31;     // fallback: one 64-elem row per lane
#pragma unroll
    for (int c = 0; c < 4; ++c)
        *(v16us*)(ldst + lane * 64 + c * 16) =
            *(const v16us*)(gsrc + lane * 64 + c * 16);
#endif
}

// ---------------------------------------------------------------------------
// Pre-passes
// ---------------------------------------------------------------------------
__global__ void cvt_bf16_kernel(const float* __restrict__ src,
                                unsigned short* __restrict__ dst, int n) {
    for (int i = blockIdx.x * blockDim.x + threadIdx.x; i < n;
         i += gridDim.x * blockDim.x)
        dst[i] = bfb(src[i]);
}

// CN [N][D][R] -> CNT [N][R][D] bf16
__global__ void tr_cn_kernel(const float* __restrict__ cn,
                             unsigned short* __restrict__ cnt) {
    int idx = blockIdx.x * blockDim.x + threadIdx.x;
    if (idx >= Nn * Rn * Dn) return;
    int n = idx >> 17, r = (idx >> 10) & (Rn - 1), d = idx & (Dn - 1);
    cnt[idx] = bfb(cn[(n << 17) + d * Rn + r]);
}

// P [N][R][D] -> PT [N][D][R] bf16
__global__ void tr_pool_kernel(const float* __restrict__ p,
                               unsigned short* __restrict__ pt) {
    int idx = blockIdx.x * blockDim.x + threadIdx.x;
    if (idx >= Nn * Dn * Rn) return;
    int n = idx >> 17, d = (idx >> 7) & (Dn - 1), r = idx & (Rn - 1);
    pt[idx] = bfb(p[(n << 17) + r * Dn + d]);
}

// ---------------------------------------------------------------------------
// Compress GEMM: 32x64 tile per wave, ping-pong pipelined, K = 16384
// ---------------------------------------------------------------------------
__global__ void __launch_bounds__(256, 1)
compress_kernel(const float* __restrict__ x,
                const float* __restrict__ cw,
                const unsigned short* __restrict__ cnt,
                float* __restrict__ h) {
    const int lane = threadIdx.x & 31;
    const int w    = (blockIdx.x * blockDim.x + threadIdx.x) >> 5;   // 256 waves
    const int m0 = (w >> 1) << 5, n0 = (w & 1) << 6;
    const int col = lane & 15, half = lane >> 4;
    const int r0 = m0 + col, r1 = r0 + 16;

    v8f acc[2][4] = {};
    auto load_all = [&](int kb, Frags& F) {
        const int n = kb >> 10, d0 = kb & (Dn - 1);
        load_a_f32(F.A0, x + r0 * Dn, d0, half, cw[r0 * Nn + n]);
        load_a_f32(F.A1, x + r1 * Dn, d0, half, cw[r1 * Nn + n]);
#pragma unroll
        for (int t = 0; t < 4; ++t)
            load_b_contig(F.B[t],
                cnt + (n << 17) + (n0 + t * 16 + col) * Dn + d0 + half * 16);
    };

    constexpr int Ktot = Nn * Dn;
    Frags F0, F1;
    load_all(0, F0);
    int kb = 0;
    for (; kb < Ktot - 64; kb += 64) {
        load_all(kb + 32, F1);
        wmma8(F0, acc);
        load_all(kb + 64, F0);
        wmma8(F1, acc);
    }
    load_all(kb + 32, F1);                 // tail (kb == Ktot-64)
    wmma8(F0, acc);
    wmma8(F1, acc);

#pragma unroll
    for (int g = 0; g < 2; ++g)
#pragma unroll
        for (int t = 0; t < 4; ++t)
#pragma unroll
            for (int i = 0; i < 8; ++i)
                h[(m0 + g * 16 + i + half * 8) * Rn + n0 + t * 16 + col] =
                    acc[g][t][i];
}

// ---------------------------------------------------------------------------
// Expand GEMM: 32x64 tile per wave, ping-pong pipelined, K = 2048
// ---------------------------------------------------------------------------
__global__ void __launch_bounds__(256, 1)
expand_kernel(const float* __restrict__ h,
              const float* __restrict__ ew,
              const unsigned short* __restrict__ pt,
              unsigned short* __restrict__ out) {
    const int lane = threadIdx.x & 31;
    const int w    = (blockIdx.x * blockDim.x + threadIdx.x) >> 5;   // 2048 waves
    const int m0 = (w >> 4) << 5, n0 = (w & 15) << 6;
    const int col = lane & 15, half = lane >> 4;
    const int r0 = m0 + col, r1 = r0 + 16;

    v8f acc[2][4] = {};
    auto load_all = [&](int kb, Frags& F) {
        const int n = kb >> 7, rr = kb & (Rn - 1);
        load_a_f32(F.A0, h + r0 * Rn, rr, half, ew[r0 * Nn + n]);
        load_a_f32(F.A1, h + r1 * Rn, rr, half, ew[r1 * Nn + n]);
#pragma unroll
        for (int t = 0; t < 4; ++t)
            load_b_contig(F.B[t],
                pt + (n << 17) + (n0 + t * 16 + col) * Rn + rr + half * 16);
    };

    constexpr int Ktot = Nn * Rn;
    Frags F0, F1;
    load_all(0, F0);
    int kb = 0;
    for (; kb < Ktot - 64; kb += 64) {
        load_all(kb + 32, F1);
        wmma8(F0, acc);
        load_all(kb + 64, F0);
        wmma8(F1, acc);
    }
    load_all(kb + 32, F1);
    wmma8(F0, acc);
    wmma8(F1, acc);

#pragma unroll
    for (int g = 0; g < 2; ++g)
#pragma unroll
        for (int t = 0; t < 4; ++t)
#pragma unroll
            for (int i = 0; i < 8; ++i) {
                const int tok = m0 + g * 16 + i + half * 8;
                const int b = tok >> 11, s = tok & (Sn - 1);
                const int d = n0 + t * 16 + col;
                out[(((b * Hn + (d >> 6)) << 11) + s) * DHn + (d & 63)] =
                    bfb(acc[g][t][i]);
            }
}

// ---------------------------------------------------------------------------
// Causal flash attention: wave per (b,h,16-query tile); 32-key blocks.
// K/V tiles staged to LDS by TDM, double buffered on TENSORcnt.
// ---------------------------------------------------------------------------
__global__ void __launch_bounds__(64, 1)
attn_kernel(const unsigned short* __restrict__ Q,
            const unsigned short* __restrict__ K,
            const unsigned short* __restrict__ V,
            unsigned short* __restrict__ AO) {
    __shared__ unsigned short s_k[2][2][32 * 64];   // [wave][buf]
    __shared__ unsigned short s_v[2][2][32 * 64];
    __shared__ unsigned short s_p[2][16 * 32];

    const int lane = threadIdx.x & 31, wid = threadIdx.x >> 5;
    const int w  = blockIdx.x * 2 + wid;            // 4096 waves total
    const int qt = w & 127, bh = w >> 7;
    const int q0 = qt << 4;
    const int row = lane & 15, half = lane >> 4;

    const unsigned short* Qp = Q + (size_t)bh * Sn * DHn;
    const unsigned short* Kp = K + (size_t)bh * Sn * DHn;
    const unsigned short* Vp = V + (size_t)bh * Sn * DHn;
    unsigned short* myp = &s_p[wid][0];

    BFrag aq[2];
#pragma unroll
    for (int ks = 0; ks < 2; ++ks)
        load_a_bf(aq[ks], Qp + (q0 + row) * DHn, ks * 32, half);

    v8f o[4] = {};
    float m_run[8], l_run[8];
#pragma unroll
    for (int i = 0; i < 8; ++i) { m_run[i] = -3.0e38f; l_run[i] = 0.0f; }

    const float scale = 0.125f;                     // 1/sqrt(64)
    const int nblocks = (q0 + 16 + 31) >> 5;

    tdm_load_kv(Kp, &s_k[wid][0][0]);               // prime buffer 0
    tdm_load_kv(Vp, &s_v[wid][0][0]);

    for (int kb = 0; kb < nblocks; ++kb) {
        const int k0 = kb << 5;
        const int buf = kb & 1;
        if (kb + 1 < nblocks) {                     // prefetch next block
            tdm_load_kv(Kp + (k0 + 32) * DHn, &s_k[wid][buf ^ 1][0]);
            tdm_load_kv(Vp + (k0 + 32) * DHn, &s_v[wid][buf ^ 1][0]);
        }
#if HAVE_TDM
        if (kb + 1 < nblocks) __builtin_amdgcn_s_wait_tensorcnt(2);
        else                  __builtin_amdgcn_s_wait_tensorcnt(0);
#else
        asm volatile("s_wait_dscnt 0" ::: "memory");
#endif
        const unsigned short* Kl = &s_k[wid][buf][0];
        const unsigned short* Vl = &s_v[wid][buf][0];

        v8f sa[2] = {};
#pragma unroll
        for (int ks = 0; ks < 2; ++ks)
#pragma unroll
            for (int nt = 0; nt < 2; ++nt) {
                BFrag Bf;
                load_b_contig(Bf,
                    Kl + (nt * 16 + row) * DHn + ks * 32 + half * 16);
                sa[nt] = wmma_bf16(aq[ks], Bf, sa[nt]);
            }

        float p[2][8];
#pragma unroll
        for (int i = 0; i < 8; ++i) {
            const int r = q0 + i + half * 8;
            float mb = -3.0e38f;
#pragma unroll
            for (int nt = 0; nt < 2; ++nt) {
                const int key = k0 + nt * 16 + row;
                float s = sa[nt][i] * scale;
                if (key > r) s = -3.0e38f;          // causal mask
                sa[nt][i] = s;
                mb = fmaxf(mb, s);
            }
#pragma unroll
            for (int d = 1; d < 16; d <<= 1)
                mb = fmaxf(mb, __shfl_xor(mb, d, 32));
            const float mn = fmaxf(m_run[i], mb);
            const float corr = __expf(m_run[i] - mn);
            m_run[i] = mn;
            float rs = 0.0f;
#pragma unroll
            for (int nt = 0; nt < 2; ++nt) {
                const float pe = __expf(sa[nt][i] - mn);
                p[nt][i] = pe; rs += pe;
            }
#pragma unroll
            for (int d = 1; d < 16; d <<= 1) rs += __shfl_xor(rs, d, 32);
            l_run[i] = l_run[i] * corr + rs;
#pragma unroll
            for (int t = 0; t < 4; ++t) o[t][i] *= corr;
        }

        // C-layout -> A-layout transpose of P through per-wave LDS
#pragma unroll
        for (int i = 0; i < 8; ++i) {
            const int rl = i + half * 8;
#pragma unroll
            for (int nt = 0; nt < 2; ++nt)
                myp[rl * 32 + nt * 16 + row] = bfb(p[nt][i]);
        }
        asm volatile("s_wait_dscnt 0" ::: "memory");

        BFrag pf;
#pragma unroll
        for (int e = 0; e < 8; ++e) {
            pf.u[e]     = myp[row * 32 + half * 8 + e];
            pf.u[e + 8] = myp[row * 32 + 16 + half * 8 + e];
        }

#pragma unroll
        for (int t = 0; t < 4; ++t) {
            const int dh = t * 16 + row;
            BFrag Bf;
#pragma unroll
            for (int e = 0; e < 16; ++e)
                Bf.u[e] = Vl[((e & 7) + half * 8 + (e >> 3) * 16) * DHn + dh];
            o[t] = wmma_bf16(pf, Bf, o[t]);
        }
    }

    const int b = bh >> 4, hh = bh & 15;
#pragma unroll
    for (int i = 0; i < 8; ++i) {
        const float inv = 1.0f / l_run[i];
        const int s = q0 + i + half * 8;
#pragma unroll
        for (int t = 0; t < 4; ++t)
            AO[((size_t)b * Sn + s) * Dn + hh * DHn + t * 16 + row] =
                bfb(o[t][i] * inv);
    }
}

// ---------------------------------------------------------------------------
// Output projection: y = AO @ W_O^T (32x64 tile per wave, ping-pong, K=1024)
// ---------------------------------------------------------------------------
__global__ void __launch_bounds__(256, 1)
final_kernel(const unsigned short* __restrict__ AO,
             const unsigned short* __restrict__ WO,
             float* __restrict__ y) {
    const int lane = threadIdx.x & 31;
    const int w    = (blockIdx.x * blockDim.x + threadIdx.x) >> 5;   // 2048 waves
    const int m0 = (w >> 4) << 5, n0 = (w & 15) << 6;
    const int col = lane & 15, half = lane >> 4;
    const int r0 = m0 + col, r1 = r0 + 16;

    v8f acc[2][4] = {};
    auto load_all = [&](int kb, Frags& F) {
        load_a_bf(F.A0, AO + (size_t)r0 * Dn, kb, half);
        load_a_bf(F.A1, AO + (size_t)r1 * Dn, kb, half);
#pragma unroll
        for (int t = 0; t < 4; ++t)
            load_b_contig(F.B[t], WO + (n0 + t * 16 + col) * Dn + kb + half * 16);
    };

    constexpr int Ktot = Dn;
    Frags F0, F1;
    load_all(0, F0);
    int kb = 0;
    for (; kb < Ktot - 64; kb += 64) {
        load_all(kb + 32, F1);
        wmma8(F0, acc);
        load_all(kb + 64, F0);
        wmma8(F1, acc);
    }
    load_all(kb + 32, F1);
    wmma8(F0, acc);
    wmma8(F1, acc);

#pragma unroll
    for (int g = 0; g < 2; ++g)
#pragma unroll
        for (int t = 0; t < 4; ++t)
#pragma unroll
            for (int i = 0; i < 8; ++i)
                y[(size_t)(m0 + g * 16 + i + half * 8) * Dn + n0 + t * 16 + col] =
                    acc[g][t][i];
}

// ---------------------------------------------------------------------------
extern "C" void kernel_launch(void* const* d_in, const int* in_sizes, int n_in,
                              void* d_out, int out_size, void* d_ws, size_t ws_size,
                              hipStream_t stream) {
    const float* x   = (const float*)d_in[0];
    const float* cw  = (const float*)d_in[1];
    const float* ewQ = (const float*)d_in[2];
    const float* ewK = (const float*)d_in[3];
    const float* ewV = (const float*)d_in[4];
    const float* cn  = (const float*)d_in[5];
    const float* pQK = (const float*)d_in[6];
    const float* pV  = (const float*)d_in[7];
    const float* wo  = (const float*)d_in[8];
    float* y = (float*)d_out;

    char* ws = (char*)d_ws;
    float*          h    = (float*)         (ws + 0);                    //  2 MB
    unsigned short* cnt  = (unsigned short*)(ws + (2u  << 20));          //  4 MB
    unsigned short* pqkt = (unsigned short*)(ws + (6u  << 20));          //  4 MB
    unsigned short* pvt  = (unsigned short*)(ws + (10u << 20));          //  4 MB
    unsigned short* wob  = (unsigned short*)(ws + (14u << 20));          //  2 MB
    unsigned short* Qb   = (unsigned short*)(ws + (16u << 20));          //  8 MB
    unsigned short* Kb   = (unsigned short*)(ws + (24u << 20));          //  8 MB
    unsigned short* Vb   = (unsigned short*)(ws + (32u << 20));          //  8 MB
    unsigned short* AO   = (unsigned short*)(ws + (40u << 20));          //  8 MB

    cvt_bf16_kernel<<<512, 256, 0, stream>>>(wo, wob, Dn * Dn);
    tr_cn_kernel  <<<(Nn * Rn * Dn + 255) / 256, 256, 0, stream>>>(cn, cnt);
    tr_pool_kernel<<<(Nn * Dn * Rn + 255) / 256, 256, 0, stream>>>(pQK, pqkt);
    tr_pool_kernel<<<(Nn * Dn * Rn + 255) / 256, 256, 0, stream>>>(pV, pvt);

    compress_kernel<<<32, 256, 0, stream>>>(x, cw, cnt, h);

    expand_kernel<<<256, 256, 0, stream>>>(h, ewQ, pqkt, Qb);
    expand_kernel<<<256, 256, 0, stream>>>(h, ewK, pqkt, Kb);
    expand_kernel<<<256, 256, 0, stream>>>(h, ewV, pvt,  Vb);

    attn_kernel<<<2048, 64, 0, stream>>>(Qb, Kb, Vb, AO);

    final_kernel<<<256, 256, 0, stream>>>(AO, wob, y);
}
